// RSBlock_45896020525510
// MI455X (gfx1250) — matrix operand order repaired and checked by reading, not compiled, not run on hardware
//
#include <hip/hip_runtime.h>
#include <stdint.h>

// ---------------------------------------------------------------------------
// MI455X (gfx1250) fused implementation.
//   prep_kernel : fold BN -> scale/shift, swizzle weights to bf16 B-fragments
//   fr_kernel   : FR branch (6->128->256) via v_wmma_f32_16x16x32_bf16
//   fused_kernel: H-layer + M branch (12->128->256) + gate + max-over-K
// ---------------------------------------------------------------------------

typedef __attribute__((ext_vector_type(16))) __bf16 v16bf;
typedef __attribute__((ext_vector_type(8)))  __bf16 v8bf;
typedef __attribute__((ext_vector_type(8)))  float  v8f;

#define BATCH   8
#define NPTS    4096
#define NROWS   (BATCH * NPTS)         /* 32768 */
#define KWIN    9
#define KSH     4
#define CLIGHT  0.49965409666666664f   /* 299792458/1e6/600 */
#define BNEPS   1e-3f

/* ---- workspace layout (bytes); total ~33.7 MB ---- */
#define FR_OFF    0
#define FR_BYTES  (NROWS * 256 * 4)              /* 33554432 */
#define MW1S_OFF  (FR_OFF + FR_BYTES)            /* 65536 B  */
#define FW1S_OFF  (MW1S_OFF + 65536)             /* 65536 B  */
#define MW0S_OFF  (FW1S_OFF + 65536)             /* 8192 B   */
#define FW0S_OFF  (MW0S_OFF + 8192)              /* 8192 B   */
#define BN_OFF    (FW0S_OFF + 8192)              /* 1536 f32 */

/* bn float offsets inside BN region */
#define BN_MS0 0
#define BN_MH0 128
#define BN_MS1 256
#define BN_MH1 512
#define BN_FS0 768
#define BN_FH0 896
#define BN_FS1 1024
#define BN_FH1 1280

/* Native f32->bf16 conversion (RTNE fptrunc -> v_cvt_*bf16_f32 on gfx1250).
   Avoids the 5-op manual rounding sequence seen in round-1 disassembly. */
__device__ __forceinline__ __bf16 to_bf16(float f) {
  return (__bf16)f;
}

/* Load one 16x32 bf16 A-fragment from LDS laid out row-major with
   `strideH` halfs per row.  ISA A-layout (16-bit A 16x32):
   lanes 0-15 row M=lane, k = {half*8+j , 16+half*8+j}.  Two b128 loads. */
__device__ __forceinline__ v16bf load_a_frag(const __bf16* base, int row,
                                             int strideH, int kOff, int half) {
  const v8bf* p = (const v8bf*)(base + row * strideH + kOff);
  v8bf lo = p[half];
  v8bf hi = p[2 + half];
  return __builtin_shufflevector(lo, hi, 0,1,2,3,4,5,6,7,8,9,10,11,12,13,14,15);
}

// ---------------------------------------------------------------------------
// prep: BN folding + weight swizzle into B-fragment order.
// B-layout (16-bit B KxN): lane = column (lanes 0-15 / repeated 16-31),
// element j -> k = (lane>>4)*8 + j (j<8), 16 + (lane>>4)*8 + (j-8) (j>=8).  Stored so each lane reads a
// contiguous v16bf:  Ws[(tile*32 + lane)*16 + j].
// ---------------------------------------------------------------------------
__global__ void __launch_bounds__(256) prep_kernel(
    const float* __restrict__ mW0,  const float* __restrict__ fW0,
    const float* __restrict__ mW1,  const float* __restrict__ fW1,
    const float* __restrict__ m_g0, const float* __restrict__ f_g0,
    const float* __restrict__ m_be0,const float* __restrict__ f_be0,
    const float* __restrict__ m_mu0,const float* __restrict__ f_mu0,
    const float* __restrict__ m_var0,const float* __restrict__ f_var0,
    const float* __restrict__ m_g1, const float* __restrict__ f_g1,
    const float* __restrict__ m_be1,const float* __restrict__ f_be1,
    const float* __restrict__ m_mu1,const float* __restrict__ f_mu1,
    const float* __restrict__ m_var1,const float* __restrict__ f_var1,
    __bf16* __restrict__ mW1s, __bf16* __restrict__ fW1s,
    __bf16* __restrict__ mW0s, __bf16* __restrict__ fW0s,
    float* __restrict__ bn)
{
  int t = blockIdx.x * 256 + threadIdx.x;
  if (t < 65536) {                                   /* W1 (128x256) swizzle */
    const float* W  = (t < 32768) ? mW1  : fW1;
    __bf16*      Ws = (t < 32768) ? mW1s : fW1s;
    int t2 = t & 32767;
    int j = t2 & 15, lane = (t2 >> 4) & 31, tile = t2 >> 9;  /* tile=kt*16+ct */
    int kt = tile >> 4, ct = tile & 15;
    int k   = (j < 8) ? ((lane >> 4) * 8 + j) : (16 + (lane >> 4) * 8 + (j - 8));
    int col = ct * 16 + (lane & 15);
    Ws[t2] = to_bf16(W[(kt * 32 + k) * 256 + col]);
  } else if (t < 73728) {                            /* W0 (pad K->32) swizzle */
    int t2 = t - 65536;
    const float* W  = (t2 < 4096) ? mW0  : fW0;
    __bf16*      Ws = (t2 < 4096) ? mW0s : fW0s;
    int kmax        = (t2 < 4096) ? 12 : 6;
    int t3 = t2 & 4095;
    int j = t3 & 15, lane = (t3 >> 4) & 31, ct = t3 >> 9;    /* ct 0..7 */
    int k   = (j < 8) ? ((lane >> 4) * 8 + j) : (16 + (lane >> 4) * 8 + (j - 8));
    int col = ct * 16 + (lane & 15);
    Ws[t3] = (k < kmax) ? to_bf16(W[k * 128 + col]) : to_bf16(0.0f);
  } else if (t < 74496) {                            /* BN -> scale/shift */
    int t2 = t - 73728;
    if (t2 < 128) {
      int i = t2;        float s = m_g0[i] * rsqrtf(m_var0[i] + BNEPS);
      bn[BN_MS0 + i] = s; bn[BN_MH0 + i] = m_be0[i] - m_mu0[i] * s;
    } else if (t2 < 384) {
      int i = t2 - 128;  float s = m_g1[i] * rsqrtf(m_var1[i] + BNEPS);
      bn[BN_MS1 + i] = s; bn[BN_MH1 + i] = m_be1[i] - m_mu1[i] * s;
    } else if (t2 < 512) {
      int i = t2 - 384;  float s = f_g0[i] * rsqrtf(f_var0[i] + BNEPS);
      bn[BN_FS0 + i] = s; bn[BN_FH0 + i] = f_be0[i] - f_mu0[i] * s;
    } else {
      int i = t2 - 512;  float s = f_g1[i] * rsqrtf(f_var1[i] + BNEPS);
      bn[BN_FS1 + i] = s; bn[BN_FH1 + i] = f_be1[i] - f_mu1[i] * s;
    }
  }
}

// ---------------------------------------------------------------------------
// FR branch: 16 rows/block, 8 waves. L0: one 16-col tile per wave.
// L1: two 16-col tiles per wave, 4 k-steps each.  fr stored f32 in ws.
// ---------------------------------------------------------------------------
__global__ void __launch_bounds__(256) fr_kernel(
    const float* __restrict__ x,
    const __bf16* __restrict__ fW0s, const __bf16* __restrict__ fW1s,
    const float* __restrict__ f_b0,  const float* __restrict__ f_b1,
    const float* __restrict__ bn,
    float* __restrict__ fr)
{
  const int row0 = blockIdx.x * 16;
  const int tid  = threadIdx.x;
  const int lane = tid & 31;
  const int wave = tid >> 5;
  const int half = lane >> 4;
  const int lr   = lane & 15;

  __shared__ float xs[16 * 6];
  __shared__ __align__(16) __bf16 y0[16 * 136];   /* padded stride: bank-spread */

  if (tid < 96) xs[tid] = x[row0 * 6 + tid];
  __syncthreads();

  /* ---- layer0: [16x6(->32)] x [32x128] ---- */
  {
    v16bf a;
    #pragma unroll
    for (int j = 0; j < 16; ++j) {
      int k = (j < 8) ? (half * 8 + j) : (16 + half * 8 + (j - 8));
      a[j] = (k < 6) ? to_bf16(xs[lr * 6 + k]) : to_bf16(0.0f);
    }
    v16bf b = ((const v16bf*)fW0s)[wave * 32 + lane];
    v8f acc = {};
    acc = __builtin_amdgcn_wmma_f32_16x16x32_bf16(false, a, false, b,
                                                  (short)0, acc, false, false);
    int col = wave * 16 + lr;
    float bias = f_b0[col], sc = bn[BN_FS0 + col], sh = bn[BN_FH0 + col];
    #pragma unroll
    for (int i = 0; i < 8; ++i) {
      int m = i + 8 * half;
      y0[m * 136 + col] = to_bf16(fmaxf(acc[i] + bias, 0.0f) * sc + sh);
    }
  }
  __syncthreads();

  /* ---- layer1: [16x128] x [128x256] ---- */
  #pragma unroll
  for (int cc = 0; cc < 2; ++cc) {
    int ct = wave + cc * 8;
    v8f acc = {};
    #pragma unroll
    for (int kt = 0; kt < 4; ++kt) {
      v16bf a = load_a_frag(y0, lr, 136, kt * 32, half);
      v16bf b = ((const v16bf*)fW1s)[(kt * 16 + ct) * 32 + lane];
      acc = __builtin_amdgcn_wmma_f32_16x16x32_bf16(false, a, false, b,
                                                    (short)0, acc, false, false);
    }
    int col = ct * 16 + lr;
    float bias = f_b1[col], sc = bn[BN_FS1 + col], sh = bn[BN_FH1 + col];
    #pragma unroll
    for (int i = 0; i < 8; ++i) {
      int m = i + 8 * half;
      fr[(row0 + m) * 256 + col] = fmaxf(acc[i] + bias, 0.0f) * sc + sh;
    }
  }
}

// ---------------------------------------------------------------------------
// Fused H + M branch + gate + max-over-K.  16 n-values per block.
// M-GEMM rows flattened row = k*16 + n_local  ->  each 16-row WMMA tile is a
// fixed k, so max-over-K is elementwise across 9 D fragments.
// ---------------------------------------------------------------------------
__global__ void __launch_bounds__(256) fused_kernel(
    const float* __restrict__ x,
    const float* __restrict__ fr,
    const __bf16* __restrict__ mW0s, const __bf16* __restrict__ mW1s,
    const float* __restrict__ m_b0,  const float* __restrict__ m_b1,
    const float* __restrict__ bn,
    float* __restrict__ out)
{
  const int b    = blockIdx.x >> 8;            /* NPTS/16 = 256 blocks per b */
  const int n0   = (blockIdx.x & 255) * 16;
  const int tid  = threadIdx.x;
  const int lane = tid & 31;
  const int wave = tid >> 5;
  const int half = lane >> 4;
  const int lr   = lane & 15;

  /* LDS carve: y0 [144][136]bf16 (39168 B) then a reused region:
     phase1: xs[24][6]f32 (576) + hA[144][40]bf16 (11520)
     phase2: frt[24][260]f32 (24960)                     total 64128 B */
  __shared__ __align__(16) unsigned char smem[39168 + 24960];
  __bf16* y0  = (__bf16*)smem;
  float*  xs  = (float*)(smem + 39168);
  __bf16* hA  = (__bf16*)(smem + 39168 + 576);
  float*  frt = (float*)(smem + 39168);

  /* -- load x halo: rows n0-4 .. n0+19, 6 feats, zero-padded -- */
  if (tid < 144) {
    int r = tid / 6, f = tid % 6;
    int n = n0 + r - KSH;
    xs[tid] = (n >= 0 && n < NPTS) ? x[(b * NPTS + n) * 6 + f] : 0.0f;
  }
  __syncthreads();

  /* -- H-layer: build [144][12] features, bf16, padded to K=32 -- */
  if (tid < 144) {
    int k = tid >> 4, nl = tid & 15;
    int srow = nl + KSH, nrow = nl + k;
    int nidx = n0 + nl + k - KSH;
    bool valid = (nidx >= 0) && (nidx < NPTS);
    float nt  = xs[nrow * 6 + 0], nc  = xs[nrow * 6 + 1], na  = xs[nrow * 6 + 2];
    float npx = xs[nrow * 6 + 3], npy = xs[nrow * 6 + 4], npz = xs[nrow * 6 + 5];
    float st  = xs[srow * 6 + 0];
    float spx = xs[srow * 6 + 3], spy = xs[srow * 6 + 4], spz = xs[srow * 6 + 5];
    float dt = valid ? (nt  - st ) : 0.0f;
    float dx = valid ? (npx - spx) : 0.0f;
    float dy = valid ? (npy - spy) : 0.0f;
    float dz = valid ? (npz - spz) : 0.0f;
    float eu2 = dx * dx + dy * dy + dz * dz;
    float cdt = CLIGHT * dt;
    __bf16* h = hA + tid * 40;
    h[0]  = to_bf16(valid ? nt  : 0.0f);
    h[1]  = to_bf16(valid ? npx : 0.0f);
    h[2]  = to_bf16(valid ? npy : 0.0f);
    h[3]  = to_bf16(valid ? npz : 0.0f);
    h[4]  = to_bf16(valid ? nc  : 0.0f);
    h[5]  = to_bf16(valid ? na  : 0.0f);
    h[6]  = to_bf16(dt);
    h[7]  = to_bf16(dx);
    h[8]  = to_bf16(dy);
    h[9]  = to_bf16(dz);
    h[10] = to_bf16(sqrtf(eu2));
    h[11] = to_bf16(cdt * cdt - eu2);
    #pragma unroll
    for (int f = 12; f < 32; ++f) h[f] = to_bf16(0.0f);
  }
  __syncthreads();

  /* -- M layer0: 9 row-tiles x 8 col-tiles = 72 wmma, 9 per wave -- */
  for (int w = wave; w < 72; w += 8) {
    int rt = w >> 3, ct0 = w & 7;
    v16bf a = load_a_frag(hA, rt * 16 + lr, 40, 0, half);
    v16bf bfr = ((const v16bf*)mW0s)[ct0 * 32 + lane];
    v8f acc = {};
    acc = __builtin_amdgcn_wmma_f32_16x16x32_bf16(false, a, false, bfr,
                                                  (short)0, acc, false, false);
    int col = ct0 * 16 + lr;
    float bias = m_b0[col], sc = bn[BN_MS0 + col], sh = bn[BN_MH0 + col];
    #pragma unroll
    for (int i = 0; i < 8; ++i) {
      int m = i + 8 * half;
      y0[(rt * 16 + m) * 136 + col] =
          to_bf16(fmaxf(acc[i] + bias, 0.0f) * sc + sh);
    }
  }
  __syncthreads();

  /* -- stage fr halo tile (overwrites xs/hA region) -- */
  for (int idx = tid; idx < 24 * 256; idx += 256) {
    int r = idx >> 8, c = idx & 255;
    int n = n0 + r - KSH;
    frt[r * 260 + c] = (n >= 0 && n < NPTS) ? fr[(b * NPTS + n) * 256 + c]
                                            : 0.0f;
  }
  __syncthreads();

  /* -- M layer1 + gate + max over K: 2 col-tiles per wave -- */
  #pragma unroll
  for (int cc = 0; cc < 2; ++cc) {
    int ct = wave + cc * 8;
    v16bf bfr[4];
    #pragma unroll
    for (int kt = 0; kt < 4; ++kt)
      bfr[kt] = ((const v16bf*)mW1s)[(kt * 16 + ct) * 32 + lane];

    int col = ct * 16 + lr;
    float bias = m_b1[col], sc = bn[BN_MS1 + col], sh = bn[BN_MH1 + col];
    float ninf = -__builtin_inff();
    v8f best = {ninf, ninf, ninf, ninf, ninf, ninf, ninf, ninf};

    for (int k = 0; k < KWIN; ++k) {
      v8f acc = {};
      #pragma unroll
      for (int kt = 0; kt < 4; ++kt) {
        v16bf a = load_a_frag(y0, k * 16 + lr, 136, kt * 32, half);
        acc = __builtin_amdgcn_wmma_f32_16x16x32_bf16(false, a, false, bfr[kt],
                                                      (short)0, acc, false, false);
      }
      #pragma unroll
      for (int i = 0; i < 8; ++i) {
        int nl = i + 8 * half;
        float mval = fmaxf(acc[i] + bias, 0.0f) * sc + sh;
        float tv = mval * frt[(nl + k) * 260 + col];
        best[i] = fmaxf(best[i], tv);
      }
    }
    #pragma unroll
    for (int i = 0; i < 8; ++i) {
      int nl = i + 8 * half;
      out[(b * NPTS + n0 + nl) * 256 + col] = best[i];
    }
  }
}

// ---------------------------------------------------------------------------
extern "C" void kernel_launch(void* const* d_in, const int* in_sizes, int n_in,
                              void* d_out, int out_size, void* d_ws, size_t ws_size,
                              hipStream_t stream) {
  (void)in_sizes; (void)n_in; (void)out_size; (void)ws_size;
  const float* x      = (const float*)d_in[0];
  const float* mW0    = (const float*)d_in[1];
  const float* fW0    = (const float*)d_in[2];
  const float* m_b0   = (const float*)d_in[3];
  const float* f_b0   = (const float*)d_in[4];
  const float* m_g0   = (const float*)d_in[5];
  const float* f_g0   = (const float*)d_in[6];
  const float* m_be0  = (const float*)d_in[7];
  const float* f_be0  = (const float*)d_in[8];
  const float* m_mu0  = (const float*)d_in[9];
  const float* f_mu0  = (const float*)d_in[10];
  const float* m_var0 = (const float*)d_in[11];
  const float* f_var0 = (const float*)d_in[12];
  const float* mW1    = (const float*)d_in[13];
  const float* fW1    = (const float*)d_in[14];
  const float* m_b1   = (const float*)d_in[15];
  const float* f_b1   = (const float*)d_in[16];
  const float* m_g1   = (const float*)d_in[17];
  const float* f_g1   = (const float*)d_in[18];
  const float* m_be1  = (const float*)d_in[19];
  const float* f_be1  = (const float*)d_in[20];
  const float* m_mu1  = (const float*)d_in[21];
  const float* f_mu1  = (const float*)d_in[22];
  const float* m_var1 = (const float*)d_in[23];
  const float* f_var1 = (const float*)d_in[24];

  char* ws = (char*)d_ws;
  float*  fr   = (float*)(ws + FR_OFF);
  __bf16* mW1s = (__bf16*)(ws + MW1S_OFF);
  __bf16* fW1s = (__bf16*)(ws + FW1S_OFF);
  __bf16* mW0s = (__bf16*)(ws + MW0S_OFF);
  __bf16* fW0s = (__bf16*)(ws + FW0S_OFF);
  float*  bnp  = (float*)(ws + BN_OFF);

  prep_kernel<<<291, 256, 0, stream>>>(mW0, fW0, mW1, fW1,
      m_g0, f_g0, m_be0, f_be0, m_mu0, f_mu0, m_var0, f_var0,
      m_g1, f_g1, m_be1, f_be1, m_mu1, f_mu1, m_var1, f_var1,
      mW1s, fW1s, mW0s, fW0s, bnp);

  fr_kernel<<<NROWS / 16, 256, 0, stream>>>(x, fW0s, fW1s, f_b0, f_b1, bnp, fr);

  fused_kernel<<<BATCH * (NPTS / 16), 256, 0, stream>>>(
      x, fr, mW0s, mW1s, m_b0, m_b1, bnp, (float*)d_out);
}
